// PoolingModule_58523224375750
// MI455X (gfx1250) — matrix operand order repaired and checked by reading, not compiled, not run on hardware
//
#include <hip/hip_runtime.h>
#include <hip/hip_bf16.h>

// ---------------------------------------------------------------------------
// PoolingModule for MI455X (gfx1250): bf16 WMMA pipeline.
//   K0  f32->bf16 converts (x, Wq, Wkv, Wout)
//   K1  segment mean-pool + rmsnorm  -> queries_f32, qn_bf16
//   K2  q  = qn @ Wq                (WMMA bf16, f32 accum)
//   K3  kv = x  @ Wkv               (WMMA bf16, dominant 1.1 TFLOP GEMM)
//   K4  flash attention per (seg, head)  (WMMA bf16)
//   K5  r  = o  @ Wout + queries    (WMMA bf16, fused residual)
//   K6  rmsnorm -> d_out (f32)
// ---------------------------------------------------------------------------

typedef __bf16 bf16;
typedef __attribute__((ext_vector_type(16))) __bf16 v16bf;
typedef __attribute__((ext_vector_type(8)))  __bf16 v8bf;
typedef __attribute__((ext_vector_type(8)))  float  v8f;

#define D_MODEL 4096
#define N_HEADS 32
#define HEAD_D  128
#define SEG_LEN 2048
#define N_SEGS  8
#define RATE    64
#define CHUNK   32
#define N_TOK   (N_SEGS * SEG_LEN)   // 16384
#define M_Q     (N_SEGS * RATE)      // 512
#define KV_N    (2 * D_MODEL)        // 8192

// concat two 8-element bf16 vectors into a WMMA operand vector
__device__ __forceinline__ v16bf concat8(const bf16* p0, const bf16* p1) {
  v8bf lo = *(const v8bf*)p0;
  v8bf hi = *(const v8bf*)p1;
  return __builtin_shufflevector(lo, hi, 0,1,2,3,4,5,6,7,8,9,10,11,12,13,14,15);
}

// ---------------------------------------------------------------- converts
__global__ __launch_bounds__(256)
void cvt_f32_bf16(const float* __restrict__ in, bf16* __restrict__ out, long n) {
  long i = ((long)blockIdx.x * 256 + threadIdx.x) * 8;
  if (i + 8 <= n) {
    float4 a = *(const float4*)(in + i);
    float4 b = *(const float4*)(in + i + 4);
    v8bf o;
    o[0] = (bf16)a.x; o[1] = (bf16)a.y; o[2] = (bf16)a.z; o[3] = (bf16)a.w;
    o[4] = (bf16)b.x; o[5] = (bf16)b.y; o[6] = (bf16)b.z; o[7] = (bf16)b.w;
    *(v8bf*)(out + i) = o;
  }
}

// --------------------------------------------------- pool + rmsnorm(queries)
__global__ __launch_bounds__(256)
void pool_qnorm(const float* __restrict__ x, const float* __restrict__ wn,
                float* __restrict__ queries, bf16* __restrict__ qn) {
  __shared__ float red[256];
  const int b   = blockIdx.x;          // pooled row 0..511 (b = s*RATE + r)
  const int tid = threadIdx.x;
  float acc[16];
#pragma unroll
  for (int j = 0; j < 16; ++j) acc[j] = 0.f;
  const float* base = x + (long)b * CHUNK * D_MODEL;
  for (int t = 0; t < CHUNK; ++t) {
    const float* row = base + (long)t * D_MODEL;
#pragma unroll
    for (int j = 0; j < 16; ++j) acc[j] += row[tid + j * 256];
  }
  float ss = 0.f;
#pragma unroll
  for (int j = 0; j < 16; ++j) { acc[j] *= (1.0f / CHUNK); ss += acc[j] * acc[j]; }
  red[tid] = ss; __syncthreads();
  for (int o = 128; o > 0; o >>= 1) {
    if (tid < o) red[tid] += red[tid + o];
    __syncthreads();
  }
  const float inv = rsqrtf(red[0] / D_MODEL + 1e-5f);
#pragma unroll
  for (int j = 0; j < 16; ++j) {
    int col = tid + j * 256;
    float q = acc[j];
    queries[(long)b * D_MODEL + col] = q;
    qn[(long)b * D_MODEL + col] = (bf16)(q * inv * wn[col]);
  }
}

// -------------------------------------------------------------- final rmsnorm
__global__ __launch_bounds__(256)
void rmsnorm_out(const float* __restrict__ r, const float* __restrict__ wn,
                 float* __restrict__ out) {
  __shared__ float red[256];
  const int b   = blockIdx.x;
  const int tid = threadIdx.x;
  float v[16]; float ss = 0.f;
#pragma unroll
  for (int j = 0; j < 16; ++j) {
    v[j] = r[(long)b * D_MODEL + tid + j * 256];
    ss += v[j] * v[j];
  }
  red[tid] = ss; __syncthreads();
  for (int o = 128; o > 0; o >>= 1) {
    if (tid < o) red[tid] += red[tid + o];
    __syncthreads();
  }
  const float inv = rsqrtf(red[0] / D_MODEL + 1e-5f);
#pragma unroll
  for (int j = 0; j < 16; ++j) {
    int col = tid + j * 256;
    out[(long)b * D_MODEL + col] = v[j] * inv * wn[col];
  }
}

// ------------------------------------------------------------ bf16 WMMA GEMM
// C[M][N] = A[M][K] @ B[K][N].  BM=BN=128, BK=32, 256 threads = 8 waves,
// each wave computes a 64x32 patch = 4x2 v_wmma_f32_16x16x32_bf16 tiles.
// OUTMODE 0: bf16 C.  OUTMODE 1: f32 C with f32 residual add.
template <int OUTMODE>
__global__ __launch_bounds__(256)
void gemm_bf16(const bf16* __restrict__ A, const bf16* __restrict__ B,
               void* __restrict__ Cout, const float* __restrict__ resid,
               int K, int lda, int ldb, int ldc) {
  constexpr int BM = 128, BN = 128, BK = 32;
  constexpr int LSA = 40, LSB = 40;        // padded LDS pitches (bank-conflict free)
  __shared__ bf16 As[BM * LSA];            // [m][k]
  __shared__ bf16 Bs[BN * LSB];            // transposed: [n][k]

  const int tid  = threadIdx.x;
  const int wave = tid >> 5;
  const int lane = tid & 31;
  const int wrow = (wave >> 2) * 64;       // 0 or 64
  const int wcol = (wave & 3) * 32;        // 0,32,64,96
  const long blockM = (long)blockIdx.x * BM;
  const long blockN = (long)blockIdx.y * BN;

  const int frm = lane & 15;               // fragment row (M for A, N for B)
  const int kbA = (lane >> 4) << 3;        // A K-base: 0 / 8  (split pattern)
  const int kbB = (lane >> 4) << 3;        // B K-base: 0 / 8  (split pattern)

  const v8f vzero = {0.f, 0.f, 0.f, 0.f, 0.f, 0.f, 0.f, 0.f};
  v8f acc[4][2];
#pragma unroll
  for (int s = 0; s < 4; ++s)
#pragma unroll
    for (int t = 0; t < 2; ++t) acc[s][t] = vzero;

  for (int k0 = 0; k0 < K; k0 += BK) {
    // stage A tile (row major)
#pragma unroll
    for (int c = 0; c < 2; ++c) {
      int linear = (c * 256 + tid) * 8;
      int i  = linear >> 5;
      int kk = linear & 31;
      *(v8bf*)&As[i * LSA + kk] = *(const v8bf*)(A + (blockM + i) * lda + k0 + kk);
    }
    // stage B tile transposed: coalesced global reads along N, scatter to [n][k]
#pragma unroll
    for (int c = 0; c < 2; ++c) {
      int linear = c * 256 + tid;
      int kk   = linear >> 4;
      int nblk = (linear & 15) << 3;
      v8bf vb = *(const v8bf*)(B + (long)(k0 + kk) * ldb + blockN + nblk);
#pragma unroll
      for (int j = 0; j < 8; ++j) Bs[(nblk + j) * LSB + kk] = vb[j];
    }
    if (k0 + BK < K)
      __builtin_prefetch(A + (blockM + (tid >> 2)) * lda + k0 + BK, 0, 1);
    __syncthreads();

    v16bf af[4], bfr[2];
#pragma unroll
    for (int s = 0; s < 4; ++s) {
      const bf16* p = &As[(wrow + s * 16 + frm) * LSA + kbA];
      af[s] = concat8(p, p + 16);
    }
#pragma unroll
    for (int t = 0; t < 2; ++t) {
      const bf16* p = &Bs[(wcol + t * 16 + frm) * LSB + kbB];
      bfr[t] = concat8(p, p + 16);
    }
#pragma unroll
    for (int s = 0; s < 4; ++s)
#pragma unroll
      for (int t = 0; t < 2; ++t)
        acc[s][t] = __builtin_amdgcn_wmma_f32_16x16x32_bf16(
            false, af[s], false, bfr[t], (short)0, acc[s][t], false, false);
    __syncthreads();
  }

  // epilogue: C layout lane holds (m = r + 8*(lane>=16), n = lane&15)
  const int mh = (lane >> 4) << 3;
#pragma unroll
  for (int s = 0; s < 4; ++s)
#pragma unroll
    for (int t = 0; t < 2; ++t)
#pragma unroll
      for (int r = 0; r < 8; ++r) {
        long gm = blockM + wrow + s * 16 + mh + r;
        long gn = blockN + wcol + t * 16 + (lane & 15);
        if (OUTMODE == 0) {
          ((bf16*)Cout)[gm * ldc + gn] = (bf16)acc[s][t][r];
        } else {
          long idx = gm * ldc + gn;
          ((float*)Cout)[idx] = acc[s][t][r] + resid[idx];
        }
      }
}

// --------------------------------------------------- flash attention (s, h)
// block = one (segment, head); 128 threads = 4 waves; wave w owns query rows
// [16w, 16w+16).  Streams K/V over L in tiles of 128 with online softmax.
__global__ __launch_bounds__(128)
void attn_flash(const bf16* __restrict__ q, const bf16* __restrict__ kv,
                bf16* __restrict__ o) {
  constexpr int LT = 128, LDK = 136, LDP = 136;
  __shared__ bf16 Ks[LT * LDK];            // [l][d]  == Bt layout for q.k^T
  __shared__ bf16 Vt[HEAD_D * LDK];        // [d][l]  == Bt layout for P.V
  __shared__ bf16 Pst[4 * 16 * LDP];       // per-wave P re-layout staging

  const int s    = blockIdx.x >> 5;
  const int h    = blockIdx.x & 31;
  const int tid  = threadIdx.x;
  const int wave = tid >> 5;
  const int lane = tid & 31;
  const int frm  = lane & 15;
  const int kbA  = (lane >> 4) << 3;
  const int kbB  = (lane >> 4) << 3;
  const int mh   = (lane >> 4) << 3;

  // resident q A-fragments: rows s*64 + wave*16 + frm, cols h*128 + kc*32 ...
  v16bf qf[4];
  {
    const bf16* qrow = q + (long)(s * RATE + wave * 16 + frm) * D_MODEL + h * HEAD_D;
#pragma unroll
    for (int kc = 0; kc < 4; ++kc)
      qf[kc] = concat8(qrow + kc * 32 + kbA, qrow + kc * 32 + kbA + 16);
  }

  const v8f vzero = {0.f, 0.f, 0.f, 0.f, 0.f, 0.f, 0.f, 0.f};
  v8f Oacc[8];
#pragma unroll
  for (int dt = 0; dt < 8; ++dt) Oacc[dt] = vzero;
  float rmax[8], rsum[8];
#pragma unroll
  for (int r = 0; r < 8; ++r) { rmax[r] = -3.0e38f; rsum[r] = 0.f; }
  const float scl = 0.08838834764831845f;  // 1/sqrt(128)

  for (int lb = 0; lb < SEG_LEN; lb += LT) {
    // stage K tile naturally, V tile transposed
    const bf16* kbase = kv + ((long)(s * SEG_LEN + lb)) * KV_N + h * HEAD_D;
#pragma unroll
    for (int c = 0; c < 16; ++c) {
      int linear = (c * 128 + tid) * 8;
      int l = linear >> 7;
      int d = linear & 127;
      *(v8bf*)&Ks[l * LDK + d] = *(const v8bf*)(kbase + (long)l * KV_N + d);
      v8bf vv = *(const v8bf*)(kbase + (long)l * KV_N + D_MODEL + d);
#pragma unroll
      for (int j = 0; j < 8; ++j) Vt[(d + j) * LDK + l] = vv[j];
    }
    __syncthreads();

    // S = q @ k^T  (8 l-subtiles x 4 K-chunks)
    v8f Sacc[8];
#pragma unroll
    for (int nt = 0; nt < 8; ++nt) {
      Sacc[nt] = vzero;
#pragma unroll
      for (int kc = 0; kc < 4; ++kc) {
        const bf16* p = &Ks[(nt * 16 + frm) * LDK + kc * 32 + kbB];
        v16bf bfr = concat8(p, p + 16);
        Sacc[nt] = __builtin_amdgcn_wmma_f32_16x16x32_bf16(
            false, qf[kc], false, bfr, (short)0, Sacc[nt], false, false);
      }
      Sacc[nt] = Sacc[nt] * scl;
    }

    // online softmax: row stats via 16-lane butterfly reductions
#pragma unroll
    for (int r = 0; r < 8; ++r) {
      float m = Sacc[0][r];
#pragma unroll
      for (int nt = 1; nt < 8; ++nt) m = fmaxf(m, Sacc[nt][r]);
      m = fmaxf(m, __shfl_xor(m, 1));
      m = fmaxf(m, __shfl_xor(m, 2));
      m = fmaxf(m, __shfl_xor(m, 4));
      m = fmaxf(m, __shfl_xor(m, 8));
      float mnew = fmaxf(rmax[r], m);
      float corr = __expf(rmax[r] - mnew);
      rmax[r] = mnew;
      float ps = 0.f;
#pragma unroll
      for (int nt = 0; nt < 8; ++nt) {
        float pe = __expf(Sacc[nt][r] - mnew);
        Sacc[nt][r] = pe;
        ps += pe;
      }
      ps += __shfl_xor(ps, 1);
      ps += __shfl_xor(ps, 2);
      ps += __shfl_xor(ps, 4);
      ps += __shfl_xor(ps, 8);
      rsum[r] = rsum[r] * corr + ps;
#pragma unroll
      for (int dt = 0; dt < 8; ++dt) Oacc[dt][r] *= corr;
    }

    // re-layout P (C layout -> A layout) through per-wave LDS tile
    bf16* P = &Pst[wave * 16 * LDP];
#pragma unroll
    for (int nt = 0; nt < 8; ++nt)
#pragma unroll
      for (int r = 0; r < 8; ++r)
        P[(mh + r) * LDP + nt * 16 + frm] = (bf16)Sacc[nt][r];

    v16bf pf[4];
#pragma unroll
    for (int kc = 0; kc < 4; ++kc) {
      const bf16* p = &P[frm * LDP + kc * 32 + kbA];
      pf[kc] = concat8(p, p + 16);
    }
    // O += P @ V
#pragma unroll
    for (int dt = 0; dt < 8; ++dt)
#pragma unroll
      for (int kc = 0; kc < 4; ++kc) {
        const bf16* p = &Vt[(dt * 16 + frm) * LDK + kc * 32 + kbB];
        v16bf vfr = concat8(p, p + 16);
        Oacc[dt] = __builtin_amdgcn_wmma_f32_16x16x32_bf16(
            false, pf[kc], false, vfr, (short)0, Oacc[dt], false, false);
      }
    __syncthreads();
  }

#pragma unroll
  for (int dt = 0; dt < 8; ++dt)
#pragma unroll
    for (int r = 0; r < 8; ++r) {
      long orow = s * RATE + wave * 16 + mh + r;
      long ocol = h * HEAD_D + dt * 16 + frm;
      o[orow * D_MODEL + ocol] = (bf16)(Oacc[dt][r] / rsum[r]);
    }
}

// ---------------------------------------------------------------------------
extern "C" void kernel_launch(void* const* d_in, const int* in_sizes, int n_in,
                              void* d_out, int out_size, void* d_ws, size_t ws_size,
                              hipStream_t stream) {
  const float* x    = (const float*)d_in[0];
  const float* Wq   = (const float*)d_in[1];
  const float* Wkv  = (const float*)d_in[2];
  const float* Wout = (const float*)d_in[3];
  const float* wnq  = (const float*)d_in[4];
  const float* wno  = (const float*)d_in[5];

  char* ws = (char*)d_ws;
  size_t off = 0;
  auto alloc = [&](size_t bytes) -> char* {
    char* p = ws + off;
    off += (bytes + 255) & ~(size_t)255;
    return p;
  };
  bf16*  x_b     = (bf16*) alloc((size_t)N_TOK * D_MODEL * 2);
  bf16*  wq_b    = (bf16*) alloc((size_t)D_MODEL * D_MODEL * 2);
  bf16*  wkv_b   = (bf16*) alloc((size_t)D_MODEL * KV_N * 2);
  bf16*  wout_b  = (bf16*) alloc((size_t)D_MODEL * D_MODEL * 2);
  bf16*  qn_b    = (bf16*) alloc((size_t)M_Q * D_MODEL * 2);
  bf16*  q_b     = (bf16*) alloc((size_t)M_Q * D_MODEL * 2);
  float* queries = (float*)alloc((size_t)M_Q * D_MODEL * 4);
  bf16*  kv_b    = (bf16*) alloc((size_t)N_TOK * KV_N * 2);
  bf16*  o_b     = (bf16*) alloc((size_t)M_Q * D_MODEL * 2);
  float* r_f     = (float*)alloc((size_t)M_Q * D_MODEL * 4);

  // K0: converts
  cvt_f32_bf16<<<(int)((long)N_TOK * D_MODEL / 2048), 256, 0, stream>>>(x, x_b, (long)N_TOK * D_MODEL);
  cvt_f32_bf16<<<(int)((long)D_MODEL * D_MODEL / 2048), 256, 0, stream>>>(Wq, wq_b, (long)D_MODEL * D_MODEL);
  cvt_f32_bf16<<<(int)((long)D_MODEL * KV_N / 2048), 256, 0, stream>>>(Wkv, wkv_b, (long)D_MODEL * KV_N);
  cvt_f32_bf16<<<(int)((long)D_MODEL * D_MODEL / 2048), 256, 0, stream>>>(Wout, wout_b, (long)D_MODEL * D_MODEL);

  // K1: pooling + rmsnorm(q)
  pool_qnorm<<<M_Q, 256, 0, stream>>>(x, wnq, queries, qn_b);

  // K2: q = qn @ Wq   [512 x 4096 x 4096]
  gemm_bf16<0><<<dim3(M_Q / 128, D_MODEL / 128), 256, 0, stream>>>(
      qn_b, wq_b, q_b, nullptr, D_MODEL, D_MODEL, D_MODEL, D_MODEL);

  // K3: kv = x @ Wkv  [16384 x 4096 x 8192]  (dominant GEMM)
  gemm_bf16<0><<<dim3(N_TOK / 128, KV_N / 128), 256, 0, stream>>>(
      x_b, wkv_b, kv_b, nullptr, D_MODEL, D_MODEL, KV_N, KV_N);

  // K4: block-diagonal flash attention, one block per (segment, head)
  attn_flash<<<N_SEGS * N_HEADS, 128, 0, stream>>>(q_b, kv_b, o_b);

  // K5: r = o @ Wout + queries
  gemm_bf16<1><<<dim3(M_Q / 128, D_MODEL / 128), 256, 0, stream>>>(
      o_b, wout_b, r_f, queries, D_MODEL, D_MODEL, D_MODEL, D_MODEL);

  // K6: out = rmsnorm(r)
  rmsnorm_out<<<M_Q, 256, 0, stream>>>(r_f, wno, (float*)d_out);
}
